// EncoderLayer_44805098832209
// MI455X (gfx1250) — matrix operand order repaired and checked by reading, not compiled, not run on hardware
//
#include <hip/hip_runtime.h>
#include <hip/hip_bf16.h>
#include <math.h>

// ---------------------------------------------------------------------------
// Transformer encoder layer for MI455X (gfx1250), wave32 + WMMA bf16.
// All GEMMs (QKV, attention QK^T, PV, Wo, FFN) run through
// v_wmma_f32_16x16x32_bf16 with fp32 accumulation. A-tile staging uses the
// gfx1250 async global->LDS path (ASYNCcnt); next-tile prefetch via
// global_prefetch_b8.
// ---------------------------------------------------------------------------

typedef __attribute__((ext_vector_type(16))) __bf16 v16bf;
typedef __attribute__((ext_vector_type(8)))  float  v8f;
typedef __attribute__((ext_vector_type(8)))  __bf16 bf16x8;
typedef __attribute__((ext_vector_type(2)))  __bf16 bf16x2;

#define DM   768
#define DFF  3072
#define SEQ  2048
#define NB   2
#define NH   12
#define DK   64
#define NTOK (NB * SEQ)   // 4096

// ---------------------------------------------------------------------------
// gfx1250 async global->LDS copy (16B per lane), tracked by ASYNCcnt.
// LDS offset = low 32 bits of the flat shared-memory address (ISA 10.2:
// LDS aperture truncates flat address to addr[31:0]).
// ---------------------------------------------------------------------------
__device__ __forceinline__ unsigned lds_off_u32(const void* p) {
    return (unsigned)(unsigned long long)p;
}
__device__ __forceinline__ void async_copy_b128(unsigned lds_off, const void* gptr) {
    asm volatile("global_load_async_to_lds_b128 %0, %1, off"
                 :: "v"(lds_off), "v"((unsigned long long)gptr)
                 : "memory");
}
__device__ __forceinline__ void wait_async0() {
    asm volatile("s_wait_asynccnt 0x0" ::: "memory");
}

// ---------------------------------------------------------------------------
// fp32 -> bf16 conversion (weights + activations)
// ---------------------------------------------------------------------------
__global__ __launch_bounds__(256) void f2b_kernel(const float* __restrict__ in,
                                                  __bf16* __restrict__ out, int n) {
    int i = blockIdx.x * 256 + threadIdx.x;
    if (i < n) out[i] = (__bf16)in[i];
}

// ---------------------------------------------------------------------------
// WMMA fragment loaders per ISA 7.12.2 16-bit layouts.
// A (16x32, MxK): lane l (0..15) row M=l holds K={0..7,16..23}; lanes 16..31
//                 hold K={8..15,24..31}. 2 bf16 per VGPR -> packed pair loads.
// B (32x16, KxN): lane l col N=(l&15); same K order as A (K={0..7,16..23} / {8..15,24..31}).
// koff selects the 32-wide contraction slice within a deeper LDS stage.
// ---------------------------------------------------------------------------
__device__ __forceinline__ v16bf load_frag_a(const __bf16* base, int stride,
                                             int row, int lh, int koff) {
    v16bf a;
#pragma unroll
    for (int p = 0; p < 8; ++p) {
        int kk = ((p & 3) << 1) + ((p >> 2) << 4) + lh * 8 + koff;
        bf16x2 t = *(const bf16x2*)(base + (size_t)row * stride + kk);
        a[2 * p]     = t.x;
        a[2 * p + 1] = t.y;
    }
    return a;
}

// B stored K-transposed: base[col * stride + k]
__device__ __forceinline__ v16bf load_frag_bT(const __bf16* base, int stride,
                                              int col, int lh, int koff) {
    v16bf b;
#pragma unroll
    for (int p = 0; p < 8; ++p) {
        int kk = ((p & 3) << 1) + ((p >> 2) << 4) + lh * 8 + koff;
        bf16x2 t = *(const bf16x2*)(base + (size_t)col * stride + kk);
        b[2 * p]     = t.x;
        b[2 * p + 1] = t.y;
    }
    return b;
}

__device__ __forceinline__ v8f wmma_bf16(v16bf a, v16bf b, v8f c) {
    return __builtin_amdgcn_wmma_f32_16x16x32_bf16(false, a, false, b,
                                                   (short)0, c, false, false);
}

// ---------------------------------------------------------------------------
// Generic bf16 GEMM:  C[M,N] = A[M,K] @ B[K,N] + bias
// Block: 256 threads (8 waves), tile 128x128, K-stage 64 (2 inner k-steps).
// Wave grid 4x2 -> each wave does a 32x64 sub-tile = 2x4 WMMA tiles,
// 16 WMMAs per LDS stage.
// EPI: 0 = bias, store bf16;  1 = bias, store f32;  2 = bias+GELU, store bf16.
// ---------------------------------------------------------------------------
template <int EPI>
__global__ __launch_bounds__(256) void gemm_bf16_kernel(
    const __bf16* __restrict__ A, const __bf16* __restrict__ B,
    const float* __restrict__ bias, __bf16* __restrict__ Cb,
    float* __restrict__ Cf, int M, int N, int K) {

    __shared__ __bf16 As[128 * 72];   // row-major [m][k], stride 72 (pad)
    __shared__ __bf16 BsT[128 * 72];  // [n][k], stride 72

    const int tid  = threadIdx.x;
    const int lane = tid & 31;
    const int wid  = tid >> 5;
    const int lh   = lane >> 4;
    const int lr   = lane & 15;
    const int wm   = (wid & 3) * 32;   // wave row origin in block tile
    const int wn   = (wid >> 2) * 64;  // wave col origin in block tile

    const int m0 = blockIdx.y * 128;
    const int n0 = blockIdx.x * 128;

    v8f acc[2][4];
#pragma unroll
    for (int i = 0; i < 2; ++i)
#pragma unroll
        for (int j = 0; j < 4; ++j) acc[i][j] = (v8f)(0.0f);

    // cooperative-load coordinates
    const int arow = tid >> 1;          // 0..127
    const int aseg = (tid & 1) * 32;    // 0 / 32 (elements within 64-deep k)
    const int bk   = tid >> 2;          // 0..63
    const int bn   = (tid & 3) * 32;    // 0..96

    for (int kt = 0; kt < K; kt += 64) {
        // A tile 128x64: async global->LDS, 4 x 16B per thread
        {
            const __bf16* ga = A + (size_t)(m0 + arow) * K + kt + aseg;
            unsigned la = lds_off_u32(&As[arow * 72 + aseg]);
#pragma unroll
            for (int i = 0; i < 4; ++i)
                async_copy_b128(la + i * 16, ga + i * 8);
            if (kt + 64 < K) __builtin_prefetch(ga + 64, 0, 1);
        }
        // B tile 64x128, stored transposed [n][k]
        {
            const bf16x8* src = (const bf16x8*)(B + (size_t)(kt + bk) * N + n0 + bn);
            bf16x8 v0 = src[0], v1 = src[1], v2 = src[2], v3 = src[3];
#pragma unroll
            for (int i = 0; i < 8; ++i) BsT[(bn + i) * 72 + bk]      = v0[i];
#pragma unroll
            for (int i = 0; i < 8; ++i) BsT[(bn + 8 + i) * 72 + bk]  = v1[i];
#pragma unroll
            for (int i = 0; i < 8; ++i) BsT[(bn + 16 + i) * 72 + bk] = v2[i];
#pragma unroll
            for (int i = 0; i < 8; ++i) BsT[(bn + 24 + i) * 72 + bk] = v3[i];
            if (kt + 64 < K)
                __builtin_prefetch((const __bf16*)src + (size_t)64 * N, 0, 1);
        }
        wait_async0();
        __syncthreads();

#pragma unroll
        for (int ks = 0; ks < 2; ++ks) {
            v16bf af[2], bfg[4];
#pragma unroll
            for (int mt = 0; mt < 2; ++mt)
                af[mt] = load_frag_a(As, 72, wm + mt * 16 + lr, lh, ks * 32);
#pragma unroll
            for (int nt = 0; nt < 4; ++nt)
                bfg[nt] = load_frag_bT(BsT, 72, wn + nt * 16 + lr, lh, ks * 32);

#pragma unroll
            for (int mt = 0; mt < 2; ++mt)
#pragma unroll
                for (int nt = 0; nt < 4; ++nt)
                    acc[mt][nt] = wmma_bf16(af[mt], bfg[nt], acc[mt][nt]);
        }
        __syncthreads();
    }

    // Epilogue: C layout -> lane holds col n=lr, rows r + 8*lh across VGPRs
#pragma unroll
    for (int mt = 0; mt < 2; ++mt)
#pragma unroll
        for (int nt = 0; nt < 4; ++nt) {
            int col = n0 + wn + nt * 16 + lr;
            float bv = bias[col];
#pragma unroll
            for (int r = 0; r < 8; ++r) {
                int row = m0 + wm + mt * 16 + r + lh * 8;
                float v = acc[mt][nt][r] + bv;
                if (EPI == 2)
                    v = 0.5f * v * (1.0f + erff(v * 0.70710678118654752f));
                size_t idx = (size_t)row * N + col;
                if (EPI == 1) Cf[idx] = v;
                else          Cb[idx] = (__bf16)v;
            }
        }
}

// ---------------------------------------------------------------------------
// Flash attention: grid (S/128, NH, NB), 256 threads = 8 waves.
// Wave w owns q-rows [qtile + 16w, +16). KV chunk = 64 columns
// (16 WMMAs per chunk, softmax shuffles amortized 2x vs 32-chunk).
// Q,K,V laid out [B*S, DM] with head h at column h*64 (matches GEMM output).
// ctx written bf16 [B*S, DM].
// ---------------------------------------------------------------------------
__global__ __launch_bounds__(256) void attn_kernel(
    const __bf16* __restrict__ Q, const __bf16* __restrict__ Km,
    const __bf16* __restrict__ V, const int* __restrict__ mask,
    __bf16* __restrict__ ctx) {

    __shared__ __bf16 KsT[64 * 72];      // [kv][dk], stride 72
    __shared__ __bf16 VsT[64 * 72];      // [dk][kv], stride 72
    __shared__ __bf16 Ps[8 * 16 * 72];   // per-wave 16x64 P tile, stride 72

    const int tid  = threadIdx.x;
    const int lane = tid & 31;
    const int wid  = tid >> 5;
    const int lh   = lane >> 4;
    const int lr   = lane & 15;

    const int qtile = blockIdx.x * 128;
    const int h     = blockIdx.y;
    const int b     = blockIdx.z;
    const int q0    = qtile + wid * 16;          // wave's first q row
    const size_t rowbase = (size_t)b * SEQ;      // token-row offset

    // Q fragments for this wave (dk=64 -> 2 K-steps), loaded once from global
    v16bf qf[2];
#pragma unroll
    for (int ks = 0; ks < 2; ++ks) {
        v16bf a;
#pragma unroll
        for (int p = 0; p < 8; ++p) {
            int dk = ((p & 3) << 1) + ((p >> 2) << 4) + lh * 8 + ks * 32;
            bf16x2 t = *(const bf16x2*)(Q + (rowbase + q0 + lr) * DM + h * DK + dk);
            a[2 * p] = t.x;  a[2 * p + 1] = t.y;
        }
        qf[ks] = a;
    }

    v8f ao[4];
#pragma unroll
    for (int d = 0; d < 4; ++d) ao[d] = (v8f)(0.0f);
    float mrow[8], lrow[8];
#pragma unroll
    for (int r = 0; r < 8; ++r) { mrow[r] = -1e30f; lrow[r] = 0.0f; }

    const int kvr = tid >> 2;          // 0..63
    const int seg = (tid & 3) * 16;    // 0..48

    for (int kv0 = 0; kv0 < SEQ; kv0 += 64) {
        // stage K (row-major [kv][dk]) and V (transposed [dk][kv]) chunks
        {
            const __bf16* kg = Km + (rowbase + kv0 + kvr) * DM + h * DK + seg;
            bf16x8 k0 = *(const bf16x8*)kg;
            bf16x8 k1 = *(const bf16x8*)(kg + 8);
            *(bf16x8*)&KsT[kvr * 72 + seg]     = k0;
            *(bf16x8*)&KsT[kvr * 72 + seg + 8] = k1;
            const __bf16* vg = V + (rowbase + kv0 + kvr) * DM + h * DK + seg;
            bf16x8 v0 = *(const bf16x8*)vg;
            bf16x8 v1 = *(const bf16x8*)(vg + 8);
#pragma unroll
            for (int i = 0; i < 8; ++i) VsT[(seg + i) * 72 + kvr]     = v0[i];
#pragma unroll
            for (int i = 0; i < 8; ++i) VsT[(seg + 8 + i) * 72 + kvr] = v1[i];
            if (kv0 + 64 < SEQ) {
                __builtin_prefetch(kg + (size_t)64 * DM, 0, 1);
                __builtin_prefetch(vg + (size_t)64 * DM, 0, 1);
            }
        }
        __syncthreads();

        // S = Q K^T : 4 column tiles, contraction over dk=64 (2 WMMAs each)
        v8f sacc[4];
#pragma unroll
        for (int nt = 0; nt < 4; ++nt) {
            v8f s = (v8f)(0.0f);
#pragma unroll
            for (int ks = 0; ks < 2; ++ks) {
                v16bf bfrag = load_frag_bT(KsT, 72, nt * 16 + lr, lh, ks * 32);
                s = wmma_bf16(qf[ks], bfrag, s);
            }
            sacc[nt] = s;
        }

        int mv[4];
#pragma unroll
        for (int nt = 0; nt < 4; ++nt)
            mv[nt] = mask[(size_t)b * SEQ + kv0 + nt * 16 + lr];

        // online softmax per row r (row = r + 8*lh of the wave tile)
#pragma unroll
        for (int r = 0; r < 8; ++r) {
            float sv[4];
#pragma unroll
            for (int nt = 0; nt < 4; ++nt) {
                float s = sacc[nt][r] * 0.125f;   // 1/sqrt(64)
                sv[nt] = (mv[nt] == 0) ? -1e9f : s;
            }
            float mx = fmaxf(fmaxf(sv[0], sv[1]), fmaxf(sv[2], sv[3]));
#pragma unroll
            for (int off = 1; off < 16; off <<= 1)
                mx = fmaxf(mx, __shfl_xor(mx, off, 32));
            float mn = fmaxf(mrow[r], mx);
            float al = __expf(mrow[r] - mn);
            mrow[r] = mn;
            float p[4], ls = 0.0f;
#pragma unroll
            for (int nt = 0; nt < 4; ++nt) { p[nt] = __expf(sv[nt] - mn); ls += p[nt]; }
#pragma unroll
            for (int off = 1; off < 16; off <<= 1)
                ls += __shfl_xor(ls, off, 32);
            lrow[r] = lrow[r] * al + ls;
#pragma unroll
            for (int d = 0; d < 4; ++d) ao[d][r] *= al;
            int m = r + lh * 8;
#pragma unroll
            for (int nt = 0; nt < 4; ++nt)
                Ps[(wid * 16 + m) * 72 + nt * 16 + lr] = (__bf16)p[nt];
        }
        __builtin_amdgcn_wave_barrier();   // order wave-local LDS P write->read

        // O += P V : A = P (16x64), B = V^T staged (contraction kv=64)
        v16bf pf[2];
#pragma unroll
        for (int ks = 0; ks < 2; ++ks)
            pf[ks] = load_frag_a(&Ps[wid * 16 * 72], 72, lr, lh, ks * 32);
#pragma unroll
        for (int d = 0; d < 4; ++d) {
#pragma unroll
            for (int ks = 0; ks < 2; ++ks) {
                v16bf vfrag = load_frag_bT(VsT, 72, d * 16 + lr, lh, ks * 32);
                ao[d] = wmma_bf16(pf[ks], vfrag, ao[d]);
            }
        }
        __syncthreads();
    }

    // normalize and store ctx as bf16 [B*S, DM]
#pragma unroll
    for (int r = 0; r < 8; ++r) {
        float inv = 1.0f / lrow[r];
        size_t row = rowbase + q0 + r + lh * 8;
#pragma unroll
        for (int d = 0; d < 4; ++d)
            ctx[row * DM + h * DK + d * 16 + lr] = (__bf16)(ao[d][r] * inv);
    }
}

// ---------------------------------------------------------------------------
// residual add + LayerNorm (fp32), optional bf16 copy of the output.
// One block per token row (768 cols, 256 threads x 3).
// ---------------------------------------------------------------------------
__global__ __launch_bounds__(256) void add_ln_kernel(
    const float* __restrict__ y, const float* __restrict__ res,
    const float* __restrict__ g, const float* __restrict__ be,
    float* __restrict__ outf, __bf16* __restrict__ outb) {

    __shared__ float red[256];
    const int row = blockIdx.x;
    const int t   = threadIdx.x;
    const size_t base = (size_t)row * DM;

    float v[3];
    float s = 0.0f;
#pragma unroll
    for (int i = 0; i < 3; ++i) {
        v[i] = y[base + t + 256 * i] + res[base + t + 256 * i];
        s += v[i];
    }
    red[t] = s; __syncthreads();
    for (int w = 128; w > 0; w >>= 1) {
        if (t < w) red[t] += red[t + w];
        __syncthreads();
    }
    float mu = red[0] * (1.0f / DM);
    __syncthreads();

    float s2 = 0.0f;
#pragma unroll
    for (int i = 0; i < 3; ++i) { float d = v[i] - mu; s2 += d * d; }
    red[t] = s2; __syncthreads();
    for (int w = 128; w > 0; w >>= 1) {
        if (t < w) red[t] += red[t + w];
        __syncthreads();
    }
    float var = red[0] * (1.0f / DM);
    float rs  = rsqrtf(var + 1e-5f);

#pragma unroll
    for (int i = 0; i < 3; ++i) {
        int c = t + 256 * i;
        float o = (v[i] - mu) * rs * g[c] + be[c];
        outf[base + c] = o;
        if (outb) outb[base + c] = (__bf16)o;
    }
}

// ---------------------------------------------------------------------------
// host launcher
// ---------------------------------------------------------------------------
extern "C" void kernel_launch(void* const* d_in, const int* in_sizes, int n_in,
                              void* d_out, int out_size, void* d_ws, size_t ws_size,
                              hipStream_t stream) {
    const float* x   = (const float*)d_in[0];
    const int*   msk = (const int*)d_in[1];
    const float* wq  = (const float*)d_in[2];
    const float* bq  = (const float*)d_in[3];
    const float* wk  = (const float*)d_in[4];
    const float* bk  = (const float*)d_in[5];
    const float* wv  = (const float*)d_in[6];
    const float* bv  = (const float*)d_in[7];
    const float* wo  = (const float*)d_in[8];
    const float* bo  = (const float*)d_in[9];
    const float* w1  = (const float*)d_in[10];
    const float* b1  = (const float*)d_in[11];
    const float* w2  = (const float*)d_in[12];
    const float* b2  = (const float*)d_in[13];
    const float* g1  = (const float*)d_in[14];
    const float* be1 = (const float*)d_in[15];
    const float* g2  = (const float*)d_in[16];
    const float* be2 = (const float*)d_in[17];
    float* out = (float*)d_out;

    // workspace carve-up
    char* ws = (char*)d_ws;
    size_t off = 0;
    auto alloc = [&](size_t bytes) -> char* {
        char* p = ws + off;
        off = (off + bytes + 255) & ~(size_t)255;
        return p;
    };
    __bf16* xb   = (__bf16*)alloc((size_t)NTOK * DM * 2);
    __bf16* wqb  = (__bf16*)alloc((size_t)DM * DM * 2);
    __bf16* wkb  = (__bf16*)alloc((size_t)DM * DM * 2);
    __bf16* wvb  = (__bf16*)alloc((size_t)DM * DM * 2);
    __bf16* wob  = (__bf16*)alloc((size_t)DM * DM * 2);
    __bf16* w1b  = (__bf16*)alloc((size_t)DM * DFF * 2);
    __bf16* w2b  = (__bf16*)alloc((size_t)DFF * DM * 2);
    __bf16* qb   = (__bf16*)alloc((size_t)NTOK * DM * 2);
    __bf16* kb   = (__bf16*)alloc((size_t)NTOK * DM * 2);
    __bf16* vb   = (__bf16*)alloc((size_t)NTOK * DM * 2);
    __bf16* ctxb = (__bf16*)alloc((size_t)NTOK * DM * 2);
    float*  attnf= (float*) alloc((size_t)NTOK * DM * 4);
    float*  x1f  = (float*) alloc((size_t)NTOK * DM * 4);
    __bf16* x1b  = (__bf16*)alloc((size_t)NTOK * DM * 2);
    __bf16* hb   = (__bf16*)alloc((size_t)NTOK * DFF * 2);
    float*  fff  = (float*) alloc((size_t)NTOK * DM * 4);
    (void)ws_size; (void)n_in; (void)in_sizes; (void)out_size;

    auto cvt = [&](const float* src, __bf16* dst, int n) {
        f2b_kernel<<<(n + 255) / 256, 256, 0, stream>>>(src, dst, n);
    };
    cvt(x,  xb,  NTOK * DM);
    cvt(wq, wqb, DM * DM);
    cvt(wk, wkb, DM * DM);
    cvt(wv, wvb, DM * DM);
    cvt(wo, wob, DM * DM);
    cvt(w1, w1b, DM * DFF);
    cvt(w2, w2b, DFF * DM);

    dim3 blk(256);
    // QKV projections
    gemm_bf16_kernel<0><<<dim3(DM / 128, NTOK / 128), blk, 0, stream>>>(
        xb, wqb, bq, qb, nullptr, NTOK, DM, DM);
    gemm_bf16_kernel<0><<<dim3(DM / 128, NTOK / 128), blk, 0, stream>>>(
        xb, wkb, bk, kb, nullptr, NTOK, DM, DM);
    gemm_bf16_kernel<0><<<dim3(DM / 128, NTOK / 128), blk, 0, stream>>>(
        xb, wvb, bv, vb, nullptr, NTOK, DM, DM);

    // attention
    attn_kernel<<<dim3(SEQ / 128, NH, NB), blk, 0, stream>>>(qb, kb, vb, msk, ctxb);

    // output projection + residual/LN
    gemm_bf16_kernel<1><<<dim3(DM / 128, NTOK / 128), blk, 0, stream>>>(
        ctxb, wob, bo, nullptr, attnf, NTOK, DM, DM);
    add_ln_kernel<<<NTOK, blk, 0, stream>>>(attnf, x, g1, be1, x1f, x1b);

    // FFN
    gemm_bf16_kernel<2><<<dim3(DFF / 128, NTOK / 128), blk, 0, stream>>>(
        x1b, w1b, b1, hb, nullptr, NTOK, DFF, DM);
    gemm_bf16_kernel<1><<<dim3(DM / 128, NTOK / 128), blk, 0, stream>>>(
        hb, w2b, b2, nullptr, fff, NTOK, DM, DFF);
    add_ln_kernel<<<NTOK, blk, 0, stream>>>(fff, x1f, g2, be2, out, nullptr);
}
